// System2Graph_86758339379778
// MI455X (gfx1250) — compile-verified, matrix operand order checked by reading, not run on hardware
//
#include <hip/hip_runtime.h>
#include <hip/hip_bf16.h>
#include <math.h>

// ---------------------------------------------------------------------------
// Problem: B=8, T=2048, D=1024. GRU-like recurrence:
//   h   = gelu([x_t, mem] @ W1^T + b1)       (8 x 2048)
//   p   = h @ W2^T + b2                      (8 x 1024)
//   g   = sigmoid(p); mem = mem + g*(p-mem); out[:,t,:] = mem
// Persistent kernel, fp16 WMMA, TDM staging, grid barrier per phase.
// ---------------------------------------------------------------------------

#define Bsz   8
#define Tlen  2048
#define Dim   1024
#define K1    2048           // 2*D
#define NWG   64
#define BLOCK 256            // 8 waves (wave32)

typedef _Float16 v16h __attribute__((ext_vector_type(16)));
typedef _Float16 v4h  __attribute__((ext_vector_type(4)));
typedef float    v8f  __attribute__((ext_vector_type(8)));
typedef unsigned int u32x4 __attribute__((ext_vector_type(4)));
typedef int          i32x8 __attribute__((ext_vector_type(8)));
typedef int          i32x4 __attribute__((ext_vector_type(4)));

// ---------------- fp32 -> fp16 weight conversion ---------------------------
__global__ void k_cvt_f16(const float* __restrict__ src,
                          _Float16* __restrict__ dst, int n) {
    int i = blockIdx.x * blockDim.x + threadIdx.x;
    if (i < n) dst[i] = (_Float16)src[i];
}

// ---------------- state init (must run every call: ws is not re-poisoned) --
__global__ void k_init(float* __restrict__ memf, _Float16* __restrict__ memh,
                       unsigned* __restrict__ sync) {
    int i = blockIdx.x * blockDim.x + threadIdx.x;
    if (i < Bsz * Dim) { memf[i] = 0.0f; memh[i] = (_Float16)0.0f; }
    if (i == 0) { sync[0] = 0u; sync[1] = 0u; }
}

// ---------------- TDM: 1-D contiguous global -> LDS copy -------------------
// Descriptor per CDNA5 ISA 8.3/8.4: group0 = {flags, lds_addr, gaddr, type},
// group1 packs data_size / tensor_dim0..1 / tile_dim0..2 / dim0_stride.
// This toolchain exposes the 6-arg builtin (extra int32x8 group before cpol).
__device__ __forceinline__ void tdm_load_1d(unsigned lds_off,
                                            const void* gptr,
                                            unsigned nelem /* fp16 elems */) {
    unsigned long long ga = (unsigned long long)gptr;
    u32x4 g0;
    g0.x = 1u;                                   // count=1 valid descriptor
    g0.y = lds_off;                              // LDS byte address
    g0.z = (unsigned)ga;                         // global addr [31:0]
    g0.w = (unsigned)((ga >> 32) & 0x1FFFFFFull) // global addr [56:32]
         | 0x80000000u;                          // type=2 ("image")
    i32x8 g1;
    g1[0] = (int)(1u << 16);                     // data_size=1 (2 bytes)
    g1[1] = (int)((nelem & 0xFFFFu) << 16);      // tensor_dim0[15:0]
    g1[2] = (int)(((nelem >> 16) & 0xFFFFu) | (1u << 16)); // d0 hi | tensor_dim1=1
    g1[3] = (int)((nelem & 0xFFFFu) << 16);      // tile_dim0 = nelem
    g1[4] = 1;                                   // tile_dim1 = 1
    g1[5] = (int)nelem;                          // tensor_dim0_stride
    g1[6] = 0;
    g1[7] = 0;
    i32x4 z4 = {0, 0, 0, 0};
    i32x8 z8 = {0, 0, 0, 0, 0, 0, 0, 0};
    __builtin_amdgcn_tensor_load_to_lds(g0, g1, z4, z4, z8, 0);
}

// ---------------- grid-wide barrier (generation counter) -------------------
__device__ __forceinline__ void grid_barrier(unsigned* cnt, unsigned* gen) {
    __threadfence();            // make this WG's global stores agent-visible
    __syncthreads();
    if (threadIdx.x == 0) {
        unsigned g = __hip_atomic_load(gen, __ATOMIC_RELAXED,
                                       __HIP_MEMORY_SCOPE_AGENT);
        unsigned a = __hip_atomic_fetch_add(cnt, 1u, __ATOMIC_ACQ_REL,
                                            __HIP_MEMORY_SCOPE_AGENT);
        if (a == NWG - 1) {
            __hip_atomic_store(cnt, 0u, __ATOMIC_RELAXED,
                               __HIP_MEMORY_SCOPE_AGENT);
            __hip_atomic_fetch_add(gen, 1u, __ATOMIC_RELEASE,
                                   __HIP_MEMORY_SCOPE_AGENT);
        } else {
            while (__hip_atomic_load(gen, __ATOMIC_ACQUIRE,
                                     __HIP_MEMORY_SCOPE_AGENT) == g)
                __builtin_amdgcn_s_sleep(2);
        }
    }
    __syncthreads();
}

// ---------------- persistent recurrence kernel -----------------------------
__global__ __launch_bounds__(BLOCK) void
k_recurrent(const float* __restrict__ x,        // [B,T,D] f32
            const _Float16* __restrict__ W1h,   // [2048,2048]
            const _Float16* __restrict__ W2h,   // [1024,2048]
            const float* __restrict__ b1,       // [2048]
            const float* __restrict__ b2,       // [1024]
            _Float16* __restrict__ hb,          // [8,2048] f16 scratch
            float* __restrict__ memf,           // [8,1024] f32 state
            _Float16* __restrict__ memh,        // [8,1024] f16 state
            unsigned* __restrict__ sync,        // {cnt, gen}
            float* __restrict__ out)            // [B,T,D] f32
{
    __shared__ _Float16 Atok[Bsz * Dim];                 // 16 KB: token half of A
    __shared__ _Float16 Amem[Bsz * Dim];                 // 16 KB: memory half of A
    __shared__ _Float16 Ah[Bsz * K1];                    // 32 KB: h for GEMM2
    __shared__ __align__(16) float redbuf[7 * 32 * 8];   //  7 KB: K-split partials

    const int tid  = threadIdx.x;
    const int lane = tid & 31;
    const int w    = tid >> 5;           // wave 0..7
    const int wg   = blockIdx.x;         // 0..63
    const int row8 = lane & 7;           // A-matrix row (rows 8..15 duplicated)
    const int colN = lane & 15;          // B-matrix column within 16-tile
    const int khalf = (lane >> 4) << 4;  // 0 / 16: contiguous k = 16h + s, same for A and B

    const int tile1 = 2 * wg + (w >> 2); // GEMM1 N-tile: 0..127
    const int kq    = w & 3;             // GEMM1 K-quarter
    const int tile2 = wg;                // GEMM2 N-tile: 0..63

    const _Float16* w1row = W1h + (size_t)(tile1 * 16 + colN) * K1;
    const _Float16* w2row = W2h + (size_t)(tile2 * 16 + colN) * K1;

    const unsigned amem_off = (unsigned)(unsigned long long)&Amem[0];
    const unsigned ah_off   = (unsigned)(unsigned long long)&Ah[0];

    unsigned* cnt = sync;
    unsigned* gen = sync + 1;

    for (int t = 0; t < Tlen; ++t) {
        // ---- stage A: TDM pulls f16 memory tile, threads convert tokens ----
        if (w == 0) tdm_load_1d(amem_off, memh, Bsz * Dim);
        {
            const float4* xf4 = (const float4*)x;
            #pragma unroll
            for (int r = 0; r < 8; ++r) {
                int u  = r * BLOCK + tid;          // 0..2047
                int b  = u >> 8;                   // batch row
                int d4 = u & 255;                  // float4 column
                float4 v = xf4[(size_t)b * (Tlen * Dim / 4) +
                               (size_t)t * (Dim / 4) + d4];
                v4h h4 = { (_Float16)v.x, (_Float16)v.y,
                           (_Float16)v.z, (_Float16)v.w };
                *(v4h*)&Atok[b * Dim + d4 * 4] = h4;
            }
        }
        if (w == 0) __builtin_amdgcn_s_wait_tensorcnt(0);
        __syncthreads();

        // ---- GEMM1: c[16x16] += A[16x32] * W1^T[32x16], K-quarter per wave --
        v8f c = {};
        #pragma unroll
        for (int kk = 0; kk < 16; ++kk) {
            int k0 = kq * 512 + kk * 32;
            int ka = k0 + khalf;
            const _Float16* asrc = (ka < Dim) ? &Atok[row8 * Dim + ka]
                                              : &Amem[row8 * Dim + (ka - Dim)];
            v16h a = *(const v16h*)asrc;                 // ds_load_b128 x2
            v16h b = *(const v16h*)(w1row + ka);         // global_load_b128 x2
            c = __builtin_amdgcn_wmma_f32_16x16x32_f16(
                    false, a, false, b, (short)0, c, false, false);
        }
        // K-split reduction through LDS (4 waves per tile, in-WG)
        if (kq != 0) {
            float* rb = &redbuf[(((w >> 2) * 3 + (kq - 1)) * 32 + lane) * 8];
            *(float4*)(rb)     = make_float4(c[0], c[1], c[2], c[3]);
            *(float4*)(rb + 4) = make_float4(c[4], c[5], c[6], c[7]);
        }
        __syncthreads();
        if (kq == 0) {
            #pragma unroll
            for (int p = 0; p < 3; ++p) {
                const float* rb =
                    &redbuf[(((w >> 2) * 3 + p) * 32 + lane) * 8];
                float4 lo = *(const float4*)rb;
                float4 hi = *(const float4*)(rb + 4);
                c[0] += lo.x; c[1] += lo.y; c[2] += lo.z; c[3] += lo.w;
                c[4] += hi.x; c[5] += hi.y; c[6] += hi.z; c[7] += hi.w;
            }
            if (lane < 16) {                      // valid rows 0..7 live here
                float bias = b1[tile1 * 16 + lane];
                #pragma unroll
                for (int i = 0; i < 8; ++i) {
                    float v = c[i] + bias;
                    float gl = 0.5f * v * (1.0f + erff(v * 0.70710678f));
                    hb[i * K1 + tile1 * 16 + lane] = (_Float16)gl;
                }
            }
        }
        grid_barrier(cnt, gen);                    // h fully materialized

        // ---- stage h into LDS via TDM ----
        if (w == 0) {
            tdm_load_1d(ah_off, hb, Bsz * K1);
            __builtin_amdgcn_s_wait_tensorcnt(0);
        }
        __syncthreads();

        // ---- GEMM2: p[16x16] += h[16x32] * W2^T[32x16], K-eighth per wave --
        v8f c2 = {};
        #pragma unroll
        for (int kk = 0; kk < 8; ++kk) {
            int k0 = w * 256 + kk * 32;
            int ka = k0 + khalf;
            v16h a = *(const v16h*)&Ah[row8 * K1 + ka];
            v16h b = *(const v16h*)(w2row + ka);
            c2 = __builtin_amdgcn_wmma_f32_16x16x32_f16(
                     false, a, false, b, (short)0, c2, false, false);
        }
        if (w != 0) {
            float* rb = &redbuf[((w - 1) * 32 + lane) * 8];
            *(float4*)(rb)     = make_float4(c2[0], c2[1], c2[2], c2[3]);
            *(float4*)(rb + 4) = make_float4(c2[4], c2[5], c2[6], c2[7]);
        }
        __syncthreads();
        if (w == 0) {
            #pragma unroll
            for (int p = 0; p < 7; ++p) {
                const float* rb = &redbuf[(p * 32 + lane) * 8];
                float4 lo = *(const float4*)rb;
                float4 hi = *(const float4*)(rb + 4);
                c2[0] += lo.x; c2[1] += lo.y; c2[2] += lo.z; c2[3] += lo.w;
                c2[4] += hi.x; c2[5] += hi.y; c2[6] += hi.z; c2[7] += hi.w;
            }
            if (lane < 16) {                      // rows 0..7 = batch entries
                int   n     = tile2 * 16 + lane;
                float bias2 = b2[n];
                #pragma unroll
                for (int i = 0; i < 8; ++i) {
                    float p  = c2[i] + bias2;
                    float g  = 1.0f / (1.0f + expf(-p));
                    float m0 = memf[i * Dim + n];
                    float mn = m0 + g * (p - m0);
                    memf[i * Dim + n] = mn;
                    memh[i * Dim + n] = (_Float16)mn;
                    out[((size_t)i * Tlen + t) * Dim + n] = mn;
                }
            }
        }
        grid_barrier(cnt, gen);                    // memory fully updated
    }
}

// ---------------------------------------------------------------------------
extern "C" void kernel_launch(void* const* d_in, const int* in_sizes, int n_in,
                              void* d_out, int out_size, void* d_ws,
                              size_t ws_size, hipStream_t stream) {
    const float* x  = (const float*)d_in[0];   // [8,2048,1024]
    const float* W1 = (const float*)d_in[1];   // [2048,2048]
    const float* b1 = (const float*)d_in[2];   // [2048]
    const float* W2 = (const float*)d_in[3];   // [1024,2048]
    const float* b2 = (const float*)d_in[4];   // [1024]
    float* out = (float*)d_out;

    char* ws = (char*)d_ws;
    _Float16* W1h  = (_Float16*)(ws);                            // 8 MB
    _Float16* W2h  = (_Float16*)(ws + (8u << 20));               // 4 MB
    _Float16* hb   = (_Float16*)(ws + (12u << 20));              // 32 KB
    float*    memf = (float*)   (ws + (12u << 20) + (64u << 10));// 32 KB
    _Float16* memh = (_Float16*)(ws + (12u << 20) + (128u << 10));//16 KB
    unsigned* sync = (unsigned*)(ws + (12u << 20) + (192u << 10));

    k_cvt_f16<<<(2048 * 2048 + 255) / 256, 256, 0, stream>>>(W1, W1h,
                                                             2048 * 2048);
    k_cvt_f16<<<(1024 * 2048 + 255) / 256, 256, 0, stream>>>(W2, W2h,
                                                             1024 * 2048);
    k_init<<<(Bsz * Dim + 255) / 256, 256, 0, stream>>>(memf, memh, sync);
    k_recurrent<<<NWG, BLOCK, 0, stream>>>(x, W1h, W2h, b1, b2, hb,
                                           memf, memh, sync, out);
}
